// QuantizedLinear_37220186587222
// MI455X (gfx1250) — 8-bit matrix operand order repaired and checked by reading, compiled, not run on hardware
//
#include <hip/hip_runtime.h>
#include <hip/hip_bf16.h>


typedef int  v8i __attribute__((ext_vector_type(8)));
typedef int  v4i __attribute__((ext_vector_type(4)));

typedef __attribute__((address_space(1))) v4i* gv4ip;
typedef __attribute__((address_space(3))) v4i* sv4ip;

#define N_ROWS   262144
#define IN_F     256
#define OUT_F    256
#define WAVES    8

__global__ __launch_bounds__(256) void qlin_pack_kernel(
    const int* __restrict__ w, const int* __restrict__ bias,
    const float* __restrict__ in_s, const float* __restrict__ w_s,
    const float* __restrict__ out_s, const int* __restrict__ in_zp,
    signed char* __restrict__ wpack, int* __restrict__ corr,
    float* __restrict__ scale)
{
    __shared__ int red[256];
    const int o = blockIdx.x;
    const int t = threadIdx.x;
    const int v = w[o * IN_F + t];
    wpack[o * IN_F + t] = (signed char)v;
    red[t] = v;
    __syncthreads();
    #pragma unroll
    for (int s = 128; s > 0; s >>= 1) {
        if (t < s) red[t] += red[t + s];
        __syncthreads();
    }
    if (t == 0) {
        corr[o]  = bias[o] - in_zp[0] * red[0];
        scale[o] = in_s[0] * w_s[o] / out_s[0];
    }
}

__device__ __forceinline__ unsigned pack4(v4i v) {
    unsigned t0 = __builtin_amdgcn_perm((unsigned)v.y, (unsigned)v.x, 0x0C0C0400u);
    unsigned t1 = __builtin_amdgcn_perm((unsigned)v.w, (unsigned)v.z, 0x04000C0Cu);
    return t0 | t1;
}

__global__ __launch_bounds__(256) void qlin_gemm_kernel(
    const int* __restrict__ x, const signed char* __restrict__ wpack,
    const int* __restrict__ corr, const float* __restrict__ scale,
    const int* __restrict__ out_zp, float* __restrict__ out)
{
    __shared__ unsigned char smem[OUT_F * IN_F];

    const int t = threadIdx.x;

#if __has_builtin(__builtin_amdgcn_global_load_async_to_lds_b128) && \
    __has_builtin(__builtin_amdgcn_s_wait_asynccnt)
    {
        #pragma unroll
        for (int j = 0; j < 16; ++j) {
            const int dst = t * IN_F + ((((j + t) & 15)) << 4);
            __builtin_amdgcn_global_load_async_to_lds_b128(
                (gv4ip)(wpack + t * IN_F + j * 16),
                (sv4ip)(smem + dst),
                 0,  0);
        }
        __builtin_amdgcn_s_wait_asynccnt(0);
    }
#else
    {
        const v4i* src = (const v4i*)(wpack + t * IN_F);
        #pragma unroll
        for (int j = 0; j < 16; ++j) {
            const int dst = t * IN_F + ((((j + t) & 15)) << 4);
            *(v4i*)(smem + dst) = src[j];
        }
    }
#endif
    __syncthreads();

    const int wave  = t >> 5;
    const int lane  = t & 31;
    const int n16   = lane & 15;
    const int hi    = lane >> 4;
    const int rbase = (blockIdx.x * WAVES + wave) * 16;
    const float ozp = (float)out_zp[0];

    v8i a[4];
    const int* xrow = x + (size_t)(rbase + n16) * IN_F;
    #pragma unroll
    for (int kt = 0; kt < 4; ++kt) {
        #pragma unroll
        for (int p = 0; p < 4; ++p) {
            const int k0 = kt * 64 + (p >> 1) * 32 + hi * 16 + (p & 1) * 8;
            v4i w0 = __builtin_nontemporal_load((const v4i*)(xrow + k0));
            v4i w1 = __builtin_nontemporal_load((const v4i*)(xrow + k0 + 4));
            a[kt][2 * p]     = (int)pack4(w0);
            a[kt][2 * p + 1] = (int)pack4(w1);
        }
    }

    #pragma unroll 4
    for (int ct = 0; ct < 16; ++ct) {
        const int col = ct * 16 + n16;
        v8i c = {};
        #pragma unroll
        for (int kt = 0; kt < 4; ++kt) {
            const int k0   = kt * 64 + hi * 16;
            const int rot  = (col * 16) & 255;
            const int offL = col * IN_F + ((k0 + rot) & 255);
            const int offH = col * IN_F + ((k0 + 32 + rot) & 255);
            v4i blo = *(const v4i*)(smem + offL);
            v4i bhi = *(const v4i*)(smem + offH);
            v8i b;
            b[0] = blo.x; b[1] = blo.y; b[2] = blo.z; b[3] = blo.w;
            b[4] = bhi.x; b[5] = bhi.y; b[6] = bhi.z; b[7] = bhi.w;
            c = __builtin_amdgcn_wmma_i32_16x16x64_iu8(
                     true, a[kt],  true, b, c,
                     false,  false);
        }
        const int   cr = corr[col];
        const float sc = scale[col];
        #pragma unroll
        for (int j = 0; j < 8; ++j) {
            const int   row = rbase + j + hi * 8;
            float v = fmaf((float)(c[j] + cr), sc, ozp);
            v = rintf(v);
            v = fminf(fmaxf(v, -128.0f), 127.0f);
            __builtin_nontemporal_store(v, out + (size_t)row * OUT_F + col);
        }
    }
}

extern "C" void kernel_launch(void* const* d_in, const int* in_sizes, int n_in,
                              void* d_out, int out_size, void* d_ws, size_t ws_size,
                              hipStream_t stream) {
    (void)in_sizes; (void)n_in; (void)out_size; (void)ws_size;
    const int*   x     = (const int*)d_in[0];
    const int*   w     = (const int*)d_in[1];
    const int*   bias  = (const int*)d_in[2];
    const float* in_s  = (const float*)d_in[3];
    const float* w_s   = (const float*)d_in[4];
    const float* out_s = (const float*)d_in[5];
    const int*   in_zp = (const int*)d_in[6];
    const int*   out_zp= (const int*)d_in[7];

    signed char* wpack = (signed char*)d_ws;
    int*         corr  = (int*)((char*)d_ws + 65536);
    float*       scale = (float*)((char*)d_ws + 65536 + 1024);
    float*       out   = (float*)d_out;

    qlin_pack_kernel<<<OUT_F, 256, 0, stream>>>(w, bias, in_s, w_s, out_s,
                                                in_zp, wpack, corr, scale);
    qlin_gemm_kernel<<<(N_ROWS / 16) / WAVES, 256, 0, stream>>>(x, wpack, corr,
                                                                scale, out_zp, out);
}
